// DynamicGATEncoder_2035814499128
// MI455X (gfx1250) — matrix operand order repaired and checked by reading, not compiled, not run on hardware
//
#include <hip/hip_runtime.h>
#include <hip/hip_bf16.h>

typedef __attribute__((ext_vector_type(16))) _Float16 v16h;
typedef __attribute__((ext_vector_type(8)))  float    v8f;

#define IN_CH 16
#define HID   64
#define NSLOPE 0.2f

// ---------- helpers ----------
__device__ __forceinline__ unsigned fenc(float f) {
    unsigned u = __float_as_uint(f);
    return (u & 0x80000000u) ? ~u : (u | 0x80000000u);
}
__device__ __forceinline__ float fdec(unsigned u) {
    return (u & 0x80000000u) ? __uint_as_float(u & 0x7FFFFFFFu)
                             : __uint_as_float(~u);
}
__device__ __forceinline__ float lrelu(float v) { return v > 0.f ? v : NSLOPE * v; }

__global__ void fill_u32_kernel(unsigned* __restrict__ p, unsigned v, long long n) {
    long long i = (long long)blockIdx.x * blockDim.x + threadIdx.x;
    if (i < n) p[i] = v;
}

// ---------- pack W (Krows x Ncols, f32) into f16 WMMA B-fragments ----------
// fragment dword index: ((kc*nTiles + tile)*32 + lane)*8 + v
// lane = 16*hi + lo; element (v,e) holds W[kc*32 + 8*hi + 2v + e + (v>=4 ? 8 : 0)][tile*16 + lo], 0 if K OOB
__global__ void packB_kernel(const float* __restrict__ W, unsigned* __restrict__ pack,
                             int Ncols, int Krows, int nTiles, int nKc) {
    const int total = nKc * nTiles * 256;
    const int idx = blockIdx.x * blockDim.x + threadIdx.x;
    if (idx >= total) return;
    const int v    = idx & 7;
    const int lane = (idx >> 3) & 31;
    const int tile = (idx >> 8) % nTiles;
    const int kc   = (idx >> 8) / nTiles;
    const int lo = lane & 15, hi = lane >> 4;
    const int n = tile * 16 + lo;
    const int k = kc * 32 + 8 * hi + 2 * v + (v >= 4 ? 8 : 0);
    const float f0 = (k < Krows)     ? W[(size_t)k * Ncols + n]       : 0.f;
    const float f1 = (k + 1 < Krows) ? W[(size_t)(k + 1) * Ncols + n] : 0.f;
    union { _Float16 h[2]; unsigned u; } cv;
    cv.h[0] = (_Float16)f0; cv.h[1] = (_Float16)f1;
    pack[idx] = cv.u;
}

// ---------- GEMM1: h1[N,256] = f16wmma( x[N,16] , W1[16,256] ) ----------
__global__ void gemm1_kernel(const float* __restrict__ x, const unsigned* __restrict__ pB,
                             float* __restrict__ h1, int nrows) {
    const int lane = threadIdx.x & 31;
    const int wave = blockIdx.x * (blockDim.x >> 5) + (threadIdx.x >> 5);
    const int r0 = wave * 16;
    if (r0 >= nrows) return;                     // wave-uniform
    const int lo = lane & 15, hi = lane >> 4;
    int row = r0 + lo; if (row >= nrows) row = nrows - 1;

    // A fragment: 8 contiguous floats at x[row*16 + 8*hi], K 16..31 zero-padded
    const float4* xp = (const float4*)(x + (size_t)row * IN_CH + 8 * hi);
    const float4 p0 = xp[0], p1 = xp[1];
    v16h a;
    a[0] = (_Float16)p0.x; a[1] = (_Float16)p0.y; a[2] = (_Float16)p0.z; a[3] = (_Float16)p0.w;
    a[4] = (_Float16)p1.x; a[5] = (_Float16)p1.y; a[6] = (_Float16)p1.z; a[7] = (_Float16)p1.w;
#pragma unroll
    for (int i = 8; i < 16; ++i) a[i] = (_Float16)0.f;

    const v16h* Bf = (const v16h*)pB;
    if (r0 + 16 <= nrows) {                      // wave-uniform fast path: full 16-row tile
#pragma unroll 4
        for (int nt = 0; nt < 16; ++nt) {
            const v16h b = Bf[nt * 32 + lane];
            v8f c = {};
            c = __builtin_amdgcn_wmma_f32_16x16x32_f16(false, a, false, b, (short)0, c, false, false);
            float* op = h1 + (size_t)(r0 + 8 * hi) * 256 + nt * 16 + lo;
#pragma unroll
            for (int r = 0; r < 8; ++r) op[(size_t)r * 256] = c[r];
        }
    } else {                                     // tail tile (only last wave, if N%16 != 0)
        for (int nt = 0; nt < 16; ++nt) {
            const v16h b = Bf[nt * 32 + lane];
            v8f c = {};
            c = __builtin_amdgcn_wmma_f32_16x16x32_f16(false, a, false, b, (short)0, c, false, false);
            float* op = h1 + (size_t)(r0 + 8 * hi) * 256 + nt * 16 + lo;
#pragma unroll
            for (int r = 0; r < 8; ++r)
                if (r0 + 8 * hi + r < nrows) op[(size_t)r * 256] = c[r];
        }
    }
}

// ---------- GEMM2: h2[N,64] = f16wmma( h[N,64] , W2[64,64] ) ----------
__global__ void gemm2_kernel(const float* __restrict__ hin, const unsigned* __restrict__ pB,
                             float* __restrict__ h2, int nrows) {
    const int lane = threadIdx.x & 31;
    const int wave = blockIdx.x * (blockDim.x >> 5) + (threadIdx.x >> 5);
    const int r0 = wave * 16;
    if (r0 >= nrows) return;
    const int lo = lane & 15, hi = lane >> 4;
    int row = r0 + lo; if (row >= nrows) row = nrows - 1;

    // A fragments for the two K-chunks; lanes hold 4x 8 contiguous floats
    const float* rp = hin + (size_t)row * HID;
    const float4 q0 = *(const float4*)(rp + 8 * hi);
    const float4 q1 = *(const float4*)(rp + 8 * hi + 4);
    const float4 q2 = *(const float4*)(rp + 16 + 8 * hi);
    const float4 q3 = *(const float4*)(rp + 16 + 8 * hi + 4);
    const float4 q4 = *(const float4*)(rp + 32 + 8 * hi);
    const float4 q5 = *(const float4*)(rp + 32 + 8 * hi + 4);
    const float4 q6 = *(const float4*)(rp + 48 + 8 * hi);
    const float4 q7 = *(const float4*)(rp + 48 + 8 * hi + 4);
    v16h a0, a1;
    a0[0]=(_Float16)q0.x; a0[1]=(_Float16)q0.y; a0[2]=(_Float16)q0.z; a0[3]=(_Float16)q0.w;
    a0[4]=(_Float16)q1.x; a0[5]=(_Float16)q1.y; a0[6]=(_Float16)q1.z; a0[7]=(_Float16)q1.w;
    a0[8]=(_Float16)q2.x; a0[9]=(_Float16)q2.y; a0[10]=(_Float16)q2.z; a0[11]=(_Float16)q2.w;
    a0[12]=(_Float16)q3.x; a0[13]=(_Float16)q3.y; a0[14]=(_Float16)q3.z; a0[15]=(_Float16)q3.w;
    a1[0]=(_Float16)q4.x; a1[1]=(_Float16)q4.y; a1[2]=(_Float16)q4.z; a1[3]=(_Float16)q4.w;
    a1[4]=(_Float16)q5.x; a1[5]=(_Float16)q5.y; a1[6]=(_Float16)q5.z; a1[7]=(_Float16)q5.w;
    a1[8]=(_Float16)q6.x; a1[9]=(_Float16)q6.y; a1[10]=(_Float16)q6.z; a1[11]=(_Float16)q6.w;
    a1[12]=(_Float16)q7.x; a1[13]=(_Float16)q7.y; a1[14]=(_Float16)q7.z; a1[15]=(_Float16)q7.w;

    const v16h* Bf = (const v16h*)pB;            // [(kc*4 + nt)*32 + lane]
    if (r0 + 16 <= nrows) {                      // wave-uniform fast path
#pragma unroll
        for (int nt = 0; nt < 4; ++nt) {
            const v16h b0 = Bf[(0 * 4 + nt) * 32 + lane];
            const v16h b1 = Bf[(1 * 4 + nt) * 32 + lane];
            v8f c = {};
            c = __builtin_amdgcn_wmma_f32_16x16x32_f16(false, a0, false, b0, (short)0, c, false, false);
            c = __builtin_amdgcn_wmma_f32_16x16x32_f16(false, a1, false, b1, (short)0, c, false, false);
            float* op = h2 + (size_t)(r0 + 8 * hi) * HID + nt * 16 + lo;
#pragma unroll
            for (int r = 0; r < 8; ++r) op[(size_t)r * HID] = c[r];
        }
    } else {
        for (int nt = 0; nt < 4; ++nt) {
            const v16h b0 = Bf[(0 * 4 + nt) * 32 + lane];
            const v16h b1 = Bf[(1 * 4 + nt) * 32 + lane];
            v8f c = {};
            c = __builtin_amdgcn_wmma_f32_16x16x32_f16(false, a0, false, b0, (short)0, c, false, false);
            c = __builtin_amdgcn_wmma_f32_16x16x32_f16(false, a1, false, b1, (short)0, c, false, false);
            float* op = h2 + (size_t)(r0 + 8 * hi) * HID + nt * 16 + lo;
#pragma unroll
            for (int r = 0; r < 8; ++r)
                if (r0 + 8 * hi + r < nrows) op[(size_t)r * HID] = c[r];
        }
    }
}

// ---------- attention dot products: a_src/a_dst [N,H] ----------
template <int H>
__global__ void attdot_kernel(const float* __restrict__ h, const float* __restrict__ att_src,
                              const float* __restrict__ att_dst,
                              float* __restrict__ a_src, float* __restrict__ a_dst, long long n) {
    long long gid = (long long)blockIdx.x * blockDim.x + threadIdx.x;
    if (gid >= n * H) return;
    const long long node = gid / H;
    const int hd = (int)(gid % H);
    const float4* hp = (const float4*)(h + (size_t)node * (H * HID) + hd * HID);
    const float4* as = (const float4*)(att_src + hd * HID);
    const float4* ad = (const float4*)(att_dst + hd * HID);
    float s1 = 0.f, s2 = 0.f;
#pragma unroll
    for (int c = 0; c < HID / 4; ++c) {
        const float4 hv = hp[c], av = as[c], dv = ad[c];
        s1 += hv.x * av.x + hv.y * av.y + hv.z * av.z + hv.w * av.w;
        s2 += hv.x * dv.x + hv.y * dv.y + hv.z * dv.z + hv.w * dv.w;
    }
    a_src[gid] = s1; a_dst[gid] = s2;
}

__device__ __forceinline__ void edge_sd(const long long* ei, long long E, long long e,
                                        int& s, int& d) {
    if (e < E) { s = (int)ei[e]; d = (int)ei[E + e]; }
    else       { s = d = (int)(e - E); }   // self-loops appended after real edges
}

// ---------- pass 1: segment max (order-preserving uint atomicMax) ----------
template <int H>
__global__ void edge_max_kernel(const long long* __restrict__ ei, long long E, long long Etot,
                                const float* __restrict__ a_src, const float* __restrict__ a_dst,
                                unsigned* __restrict__ emax) {
    long long e = (long long)blockIdx.x * blockDim.x + threadIdx.x;
    if (e >= Etot) return;
    int s, d; edge_sd(ei, E, e, s, d);
#pragma unroll
    for (int hd = 0; hd < H; ++hd) {
        const float ev = lrelu(a_src[(size_t)s * H + hd] + a_dst[(size_t)d * H + hd]);
        atomicMax(&emax[(size_t)d * H + hd], fenc(ev));
    }
}

// ---------- pass 2: ex = exp(e - max), segment sum ----------
template <int H>
__global__ void edge_sum_kernel(const long long* __restrict__ ei, long long E, long long Etot,
                                const float* __restrict__ a_src, const float* __restrict__ a_dst,
                                const unsigned* __restrict__ emax,
                                float* __restrict__ denom, float* __restrict__ ex) {
    long long e = (long long)blockIdx.x * blockDim.x + threadIdx.x;
    if (e >= Etot) return;
    int s, d; edge_sd(ei, E, e, s, d);
#pragma unroll
    for (int hd = 0; hd < H; ++hd) {
        const float ev = lrelu(a_src[(size_t)s * H + hd] + a_dst[(size_t)d * H + hd]);
        const float xv = expf(ev - fdec(emax[(size_t)d * H + hd]));
        ex[(size_t)e * H + hd] = xv;
        atomicAdd(&denom[(size_t)d * H + hd], xv);
    }
}

// ---------- pass 3: out[dst,c] += (1/H) * sum_h alpha_h * h[src,h,c] ----------
template <int H>
__global__ void edge_msg_kernel(const long long* __restrict__ ei, long long E, long long Etot,
                                const float* __restrict__ ex, const float* __restrict__ denom,
                                const float* __restrict__ h, float* __restrict__ out) {
    long long gid = (long long)blockIdx.x * blockDim.x + threadIdx.x;
    if (gid >= Etot * HID) return;
    const long long e = gid >> 6;
    const int c = (int)(gid & 63);
    int s, d; edge_sd(ei, E, e, s, d);
    float acc = 0.f;
#pragma unroll
    for (int hd = 0; hd < H; ++hd) {
        const float alpha = ex[(size_t)e * H + hd] / (denom[(size_t)d * H + hd] + 1e-16f);
        acc += alpha * h[(size_t)s * (H * HID) + hd * HID + c];
    }
    atomicAdd(&out[(size_t)d * HID + c], acc * (1.0f / H));
}

// ---------- finalize: in-place  v = elu(v + bias[c]) ----------
__global__ void finalize_kernel(float* __restrict__ acc, const float* __restrict__ bias,
                                long long total) {
    long long i = (long long)blockIdx.x * blockDim.x + threadIdx.x;
    if (i >= total) return;
    const float v = acc[i] + bias[i & 63];
    acc[i] = v > 0.f ? v : (expf(v) - 1.f);
}

extern "C" void kernel_launch(void* const* d_in, const int* in_sizes, int n_in,
                              void* d_out, int out_size, void* d_ws, size_t ws_size,
                              hipStream_t stream) {
    (void)n_in; (void)out_size; (void)ws_size;
    const float*     x        = (const float*)d_in[0];
    const long long* ei       = (const long long*)d_in[1];   // int64 edge_index [2,E]
    const float*     W1       = (const float*)d_in[2];
    const float*     att_src1 = (const float*)d_in[3];
    const float*     att_dst1 = (const float*)d_in[4];
    const float*     b1       = (const float*)d_in[5];
    const float*     W2       = (const float*)d_in[6];
    const float*     att_src2 = (const float*)d_in[7];
    const float*     att_dst2 = (const float*)d_in[8];
    const float*     b2       = (const float*)d_in[9];

    const long long N    = in_sizes[0] / IN_CH;
    const long long E    = in_sizes[1] / 2;
    const long long Etot = E + N;            // reference appends self-loops
    float* out = (float*)d_out;

    // workspace layout (floats; offset rounded up to 8 dwords before the packed B chunks for v16h loads)
    float* ws = (float*)d_ws;
    size_t o = 0;
    float*    h1      = ws + o; o += (size_t)N * 256;
    float*    acc1    = ws + o; o += (size_t)N * HID;   // conv1 out, then ELU'd in place
    float*    h2      = ws + o; o += (size_t)N * HID;
    float*    a_src_1 = ws + o; o += (size_t)N * 4;
    float*    a_dst_1 = ws + o; o += (size_t)N * 4;
    unsigned* emax_1  = (unsigned*)(ws + o); o += (size_t)N * 4;
    float*    denom_1 = ws + o; o += (size_t)N * 4;
    float*    ex_1    = ws + o; o += (size_t)Etot * 4;
    float*    a_src_2 = ws + o; o += (size_t)N;
    float*    a_dst_2 = ws + o; o += (size_t)N;
    unsigned* emax_2  = (unsigned*)(ws + o); o += (size_t)N;
    float*    denom_2 = ws + o; o += (size_t)N;
    float*    ex_2    = ws + o; o += (size_t)Etot;
    o = (o + 7) & ~(size_t)7;
    unsigned* pB1     = (unsigned*)(ws + o); o += 16 * 256;        // 1 kc * 16 tiles
    unsigned* pB2     = (unsigned*)(ws + o); o += 2 * 4 * 256;     // 2 kc * 4 tiles

    const int TB = 256;
    const unsigned ENC_NEG_INF = 0x007FFFFFu;   // fenc(-inf)
    auto nb = [&](long long n) { return (unsigned)((n + TB - 1) / TB); };

    // pack constant B fragments (f32 -> f16 WMMA layout)
    packB_kernel<<<nb(16 * 256), TB, 0, stream>>>(W1, pB1, 256, IN_CH, 16, 1);
    packB_kernel<<<nb(2 * 4 * 256), TB, 0, stream>>>(W2, pB2, HID, HID, 4, 2);

    // ================= layer 1 (H=4) =================
    fill_u32_kernel<<<nb(N * HID), TB, 0, stream>>>((unsigned*)acc1, 0u, N * HID);
    fill_u32_kernel<<<nb(N * 4), TB, 0, stream>>>(emax_1, ENC_NEG_INF, N * 4);
    fill_u32_kernel<<<nb(N * 4), TB, 0, stream>>>((unsigned*)denom_1, 0u, N * 4);

    const long long waves1 = (N + 15) / 16;
    gemm1_kernel<<<(unsigned)((waves1 + 7) / 8), TB, 0, stream>>>(x, pB1, h1, (int)N);
    attdot_kernel<4><<<nb(N * 4), TB, 0, stream>>>(h1, att_src1, att_dst1, a_src_1, a_dst_1, N);
    edge_max_kernel<4><<<nb(Etot), TB, 0, stream>>>(ei, E, Etot, a_src_1, a_dst_1, emax_1);
    edge_sum_kernel<4><<<nb(Etot), TB, 0, stream>>>(ei, E, Etot, a_src_1, a_dst_1, emax_1,
                                                    denom_1, ex_1);
    edge_msg_kernel<4><<<nb(Etot * HID), TB, 0, stream>>>(ei, E, Etot, ex_1, denom_1, h1, acc1);
    finalize_kernel<<<nb(N * HID), TB, 0, stream>>>(acc1, b1, N * HID);

    // ================= layer 2 (H=1) =================
    fill_u32_kernel<<<nb(N * HID), TB, 0, stream>>>((unsigned*)out, 0u, N * HID);
    fill_u32_kernel<<<nb(N), TB, 0, stream>>>(emax_2, ENC_NEG_INF, N);
    fill_u32_kernel<<<nb(N), TB, 0, stream>>>((unsigned*)denom_2, 0u, N);

    gemm2_kernel<<<(unsigned)((waves1 + 7) / 8), TB, 0, stream>>>(acc1, pB2, h2, (int)N);
    attdot_kernel<1><<<nb(N), TB, 0, stream>>>(h2, att_src2, att_dst2, a_src_2, a_dst_2, N);
    edge_max_kernel<1><<<nb(Etot), TB, 0, stream>>>(ei, E, Etot, a_src_2, a_dst_2, emax_2);
    edge_sum_kernel<1><<<nb(Etot), TB, 0, stream>>>(ei, E, Etot, a_src_2, a_dst_2, emax_2,
                                                    denom_2, ex_2);
    edge_msg_kernel<1><<<nb(Etot * HID), TB, 0, stream>>>(ei, E, Etot, ex_2, denom_2, h2, out);
    finalize_kernel<<<nb(N * HID), TB, 0, stream>>>(out, b2, N * HID);
}
